// LSTM_54984171323419
// MI455X (gfx1250) — 8-bit matrix operand order repaired and checked by reading, compiled, not run on hardware
//
#include <hip/hip_runtime.h>


#define BATCH 1024
#define SEQL  512
#define INDIM 128
#define HID   256
#define NCLS  10
#define BT    32
#define NWG   (BATCH / BT)
#define THREADS 512
#define HP    272
#define GROW  1040

#define NLOG2E  (-1.4426950408889634f)
#define N2LOG2E (-2.8853900817779268f)

typedef __attribute__((ext_vector_type(16))) int   v16i;
typedef __attribute__((ext_vector_type(8)))  float v8f;


__device__ inline float u32_as_f(unsigned u) {
  union { unsigned u; float f; } v; v.u = u; return v.f;
}
__device__ inline float bf16_to_f(unsigned short u) {
  return u32_as_f(((unsigned)u) << 16);
}
__device__ inline unsigned short f_to_bf16_rne(float f) {
  union { float f; unsigned u; } v; v.f = f;
  unsigned r = v.u + 0x7FFFu + ((v.u >> 16) & 1u);
  return (unsigned short)(r >> 16);
}
__device__ inline unsigned short f_to_bf16_rz(float f) {
  union { float f; unsigned u; } v; v.f = f;
  return (unsigned short)(v.u >> 16);
}

__device__ inline unsigned f32_to_e4m3(float x) {
  union { float f; unsigned u; } v; v.f = fabsf(x);
  unsigned s = (x < 0.f) ? 0x80u : 0u;
  if (v.f >= 448.f) return s | 0x7Eu;
  int e = ((int)(v.u >> 23)) - 127;
  if (e < -6) e = -6;
  int q = (int)rintf(ldexpf(v.f, 3 - e));
  if (q >= 16) { q >>= 1; ++e; }
  if (q >= 8) {
    if (e > 8) return s | 0x7Eu;
    return s | (unsigned)(((e + 7) << 3) | (q - 8));
  }
  return s | (unsigned)q;
}

__device__ inline float sigm_exp2(float z) {
  return __builtin_amdgcn_rcpf(1.f + __builtin_amdgcn_exp2f(z));
}

__global__ __launch_bounds__(256) void pack_wh_kernel(
    const float* __restrict__ Wg, const float* __restrict__ Wi,
    const float* __restrict__ Wf, const float* __restrict__ Wo,
    unsigned* __restrict__ whp)
{
  int widx = blockIdx.x * 256 + threadIdx.x;
  int j    = widx & 15;
  int lane = (widx >> 4) & 31;
  int nt   = (widx >> 9) & 63;
  int kc   = widx >> 15;
  int ng   = nt * 16 + (lane & 15);
  const float* W = (ng < 256) ? Wg : (ng < 512) ? Wi : (ng < 768) ? Wf : Wo;
  int n = ng & 255;
  unsigned word = 0;
  #pragma unroll
  for (int b = 0; b < 4; ++b) {
    int K = kc * 128 + 32 * (j >> 2) + 16 * (lane >> 4) + 4 * (j & 3) + b;
    unsigned q = f32_to_e4m3(16.f * W[K * HID + n]);
    word |= q << (8 * b);
  }
  whp[widx] = word;
}

__global__ __launch_bounds__(256) void pack_wx_kernel(
    const float* __restrict__ Wg, const float* __restrict__ Wi,
    const float* __restrict__ Wf, const float* __restrict__ Wo,
    const float* __restrict__ bg, const float* __restrict__ bi,
    const float* __restrict__ bf, const float* __restrict__ bo,
    unsigned short* __restrict__ wx)
{
  int e = blockIdx.x * 256 + threadIdx.x;
  int d = e >> 10;
  int j = e & 1023;
  int gate = j >> 8;
  int n = j & 255;
  const float* W = (gate == 0) ? Wg : (gate == 1) ? Wi : (gate == 2) ? Wf : Wo;
  const float* B = (gate == 0) ? bg : (gate == 1) ? bi : (gate == 2) ? bf : bo;
  float preMul = (gate == 0) ? 2.f : 1.f;
  wx[e] = f_to_bf16_rne((W[d * HID + n] + B[n]) * (NLOG2E * preMul));
}

__global__ __launch_bounds__(THREADS) void lstm_scan_kernel(
    const int* __restrict__ x,
    const unsigned* __restrict__ whp,
    const unsigned short* __restrict__ wxf,
    const float* __restrict__ Whp,
    const float* __restrict__ bp,
    float* __restrict__ out)
{
  __shared__ unsigned short s_wx[BT * 1024];
  __shared__ unsigned short s_ga[BT * GROW];
  __shared__ unsigned char  s_h8[BT * HP];
  __shared__ int            s_idx[BT];

  const int tid    = (int)threadIdx.x;
  const int lane   = tid & 31;
  const int wv     = tid >> 5;
  const int gb     = (int)blockIdx.x * BT;
  const int laneLo = lane & 15;
  const int laneHi = lane >> 4;
  const int ko     = laneHi * 8;
  const int gate   = wv >> 2;

  const float preMul  = (gate == 0) ? 2.f : 1.f;
  const float postMul = preMul;
  const float postAdd = (gate == 0) ? -1.f : 0.f;
  const float sAcc    = NLOG2E * preMul * (1.f / 256.f);

  v16i bfrag[4][2];
  #pragma unroll
  for (int ntl = 0; ntl < 4; ++ntl) {
    #pragma unroll
    for (int kc = 0; kc < 2; ++kc) {
      const int* p = (const int*)whp + (((kc * 64 + (wv * 4 + ntl)) * 32 + lane) << 4);
      #pragma unroll
      for (int v = 0; v < 16; ++v) bfrag[ntl][kc][v] = p[v];
    }
  }

  for (int i = tid; i < BT * HP; i += THREADS) s_h8[i] = 0;
  float creg[16];
  #pragma unroll
  for (int q = 0; q < 16; ++q) creg[q] = 0.f;

  const int um  = tid >> 4;
  const int uj0 = (tid & 15) << 4;

  for (int t = 0; t < SEQL; ++t) {
    if (tid < BT) s_idx[tid] = x[(gb + tid) * SEQL + t];
    __syncthreads();

    {
      const uint4* src = (const uint4*)wxf;
      uint4*       dst = (uint4*)s_wx;
      #pragma unroll
      for (int i = 0; i < 8; ++i) {
        int q  = tid + i * THREADS;
        int m  = q >> 7;
        int cc = q & 127;
        dst[q] = src[s_idx[m] * 128 + cc];
      }
    }
    __syncthreads();

    #pragma unroll
    for (int mt = 0; mt < 2; ++mt) {
      v16i a0, a1;
      const unsigned* hrow = (const unsigned*)(s_h8 + (mt * 16 + laneLo) * HP);
      #pragma unroll
      for (int v = 0; v < 16; ++v) {
        int ks = ((v >> 2) << 5) + ((v & 3) << 2) + (ko << 1);
        a0[v] = (int)hrow[ks >> 2];
        a1[v] = (int)hrow[(ks >> 2) + 32];
      }
      #pragma unroll
      for (int ntl = 0; ntl < 4; ++ntl) {
        v8f acc;
        #pragma unroll
        for (int r = 0; r < 8; ++r) acc[r] = 0.f;
        acc = __builtin_amdgcn_wmma_f32_16x16x128_fp8_fp8(a0, bfrag[ntl][0], (short)0, acc, false, false);
        acc = __builtin_amdgcn_wmma_f32_16x16x128_fp8_fp8(a1, bfrag[ntl][1], (short)0, acc, false, false);
        const int j = ((wv * 4 + ntl) << 4) + laneLo;
        #pragma unroll
        for (int r = 0; r < 8; ++r) {
          int   m  = mt * 16 + r + (laneHi << 3);
          float wz = bf16_to_f(s_wx[m * 1024 + j]);
          float z  = fmaf(acc[r], sAcc, wz);
          float a  = fmaf(sigm_exp2(z), postMul, postAdd);
          s_ga[m * GROW + j] = f_to_bf16_rz(a);
        }
      }
    }
    __syncthreads();

    {
      const unsigned* gaw = (const unsigned*)s_ga;
      const int wbase = (um * GROW + uj0) >> 1;
      unsigned pk[4];
      #pragma unroll
      for (int p2 = 0; p2 < 8; ++p2) {
        unsigned g2 = gaw[wbase + p2];
        unsigned i2 = gaw[wbase + 128 + p2];
        unsigned f2 = gaw[wbase + 256 + p2];
        unsigned o2 = gaw[wbase + 384 + p2];
        #pragma unroll
        for (int h2 = 0; h2 < 2; ++h2) {
          int   q  = p2 * 2 + h2;
          float g  = u32_as_f(h2 ? (g2 & 0xFFFF0000u) : (g2 << 16));
          float ii = u32_as_f(h2 ? (i2 & 0xFFFF0000u) : (i2 << 16));
          float ff = u32_as_f(h2 ? (f2 & 0xFFFF0000u) : (f2 << 16));
          float oo = u32_as_f(h2 ? (o2 & 0xFFFF0000u) : (o2 << 16));
          float c  = g * ii + creg[q] * ff;
          creg[q]  = c;
          float h  = fmaf(2.f, sigm_exp2(c * N2LOG2E), -1.f) * oo;
          unsigned b = f32_to_e4m3(16.f * h);
          if ((q & 3) == 0) pk[q >> 2] = b;
          else              pk[q >> 2] |= b << ((q & 3) * 8);
          if (t == SEQL - 1) ((float*)s_wx)[um * HID + uj0 + q] = h;
        }
      }
      uint4 vv; vv.x = pk[0]; vv.y = pk[1]; vv.z = pk[2]; vv.w = pk[3];
      *(uint4*)(s_h8 + um * HP + uj0) = vv;
    }
    __syncthreads();
  }

  const float* hf = (const float*)s_wx;
  for (int e = tid; e < BT * NCLS; e += THREADS) {
    int m = e / NCLS, n = e - m * NCLS;
    float s = bp[n];
    #pragma unroll 4
    for (int k = 0; k < HID; ++k) s += hf[m * HID + k] * Whp[k * NCLS + n];
    out[(gb + m) * NCLS + n] = s;
  }
}

extern "C" void kernel_launch(void* const* d_in, const int* in_sizes, int n_in,
                              void* d_out, int out_size, void* d_ws, size_t ws_size,
                              hipStream_t stream) {
  (void)in_sizes; (void)n_in; (void)out_size; (void)ws_size;
  const int*   x    = (const int*)  d_in[0];
  const float* Wxg  = (const float*)d_in[1];
  const float* Whg  = (const float*)d_in[2];
  const float* Wxi  = (const float*)d_in[3];
  const float* Whi  = (const float*)d_in[4];
  const float* Wxf  = (const float*)d_in[5];
  const float* Whf  = (const float*)d_in[6];
  const float* Wxo  = (const float*)d_in[7];
  const float* Who  = (const float*)d_in[8];
  const float* Whp  = (const float*)d_in[9];
  const float* bg   = (const float*)d_in[10];
  const float* bi   = (const float*)d_in[11];
  const float* bf_  = (const float*)d_in[12];
  const float* bo   = (const float*)d_in[13];
  const float* bp   = (const float*)d_in[14];

  unsigned*       whp_pk = (unsigned*)d_ws;
  unsigned short* wx_pk  = (unsigned short*)((char*)d_ws + 262144);

  pack_wh_kernel<<<256, 256, 0, stream>>>(Whg, Whi, Whf, Who, whp_pk);
  pack_wx_kernel<<<512, 256, 0, stream>>>(Wxg, Wxi, Wxf, Wxo, bg, bi, bf_, bo, wx_pk);
  lstm_scan_kernel<<<NWG, THREADS, 0, stream>>>(x, whp_pk, wx_pk, Whp, bp, (float*)d_out);
}
